// SelfAttention_70660801954384
// MI455X (gfx1250) — matrix operand order repaired and checked by reading, not compiled, not run on hardware
//
#include <hip/hip_runtime.h>
#include <hip/hip_bf16.h>
#include <stdint.h>

typedef __attribute__((ext_vector_type(16))) __bf16 v16bf;
typedef __attribute__((ext_vector_type(8)))  float  v8f;
typedef unsigned short u16;
typedef __attribute__((ext_vector_type(8)))  u16    u16x8;
typedef __attribute__((__vector_size__(4 * sizeof(int)))) int i32x4;

union BFPack { v16bf v; u16x8 h[2]; };

static constexpr int kB   = 2;
static constexpr int kN   = 2048;
static constexpr int kDim = 1024;
static constexpr int kH   = 16;
static constexpr int kD   = 64;       // head dim
static constexpr int kE   = 3 * kDim; // 3072 qkv rows

// ---- async global->LDS staging (CDNA5), guarded; fallback = load + ds_store
#if defined(__has_builtin)
#  if __has_builtin(__builtin_amdgcn_global_load_async_to_lds_b128)
#    define HAVE_ASYNC_LDS 1
#  endif
#endif
#ifndef HAVE_ASYNC_LDS
#  define HAVE_ASYNC_LDS 0
#endif

#if HAVE_ASYNC_LDS
__device__ __forceinline__ void async_cp16(const u16* g, u16* l) {
  __builtin_amdgcn_global_load_async_to_lds_b128(
      (__attribute__((address_space(1))) i32x4*)(uintptr_t)g,  // global addr == flat addr
      (__attribute__((address_space(3))) i32x4*)l,             // addrspacecast generic->LDS
      0, 0);
}
#endif

__device__ __forceinline__ u16 f2bf(float f) {
  union { float f; unsigned int u; } c; c.f = f;
  unsigned int u = c.u;
  unsigned int r = u + 0x7FFFu + ((u >> 16) & 1u);
  return (u16)(r >> 16);
}

// ---------------------------------------------------------------- convert
__global__ void cvt_bf16(const float* __restrict__ src, u16* __restrict__ dst, int n) {
  int i = blockIdx.x * blockDim.x + threadIdx.x;
  int stride = gridDim.x * blockDim.x;
  for (; i < n; i += stride) dst[i] = f2bf(src[i]);
}

// ---------------------------------------------------------------- QKV GEMM
// C[m,e] = sum_d x[m,d] * W[e,d] + b[e];  one wave -> 16x64 tile (A reused 4x)
__global__ void __launch_bounds__(256)
qkv_gemm(const u16* __restrict__ x16, const u16* __restrict__ w16,
         const float* __restrict__ bias,
         float* __restrict__ qf, float* __restrict__ kf, u16* __restrict__ vt16) {
  const int lane = threadIdx.x & 31;
  const int gw = blockIdx.x * 8 + (threadIdx.x >> 5);
  const int mTile = gw / 48;          // 0..255  (16 rows each)
  const int eGrp  = gw % 48;          // 0..47   (64 cols each)
  const int mBase = mTile * 16;
  const int eBase = eGrp * 64;
  const int g  = lane >> 4;
  const int ln = lane & 15;

  const u16* aRow = x16 + (size_t)(mBase + ln) * kDim;
  const u16* b0 = w16 + (size_t)(eBase + ln) * kDim;
  const u16* b1 = b0 + (size_t)16 * kDim;
  const u16* b2 = b0 + (size_t)32 * kDim;
  const u16* b3 = b0 + (size_t)48 * kDim;

  v8f acc0 = {}, acc1 = {}, acc2 = {}, acc3 = {};
  for (int kd = 0; kd < kDim; kd += 32) {
    BFPack a;
    a.h[0] = *(const u16x8*)(aRow + kd + g * 16);
    a.h[1] = *(const u16x8*)(aRow + kd + g * 16 + 8);
    if (kd + 64 < kDim) {                       // warm the streams (global_prefetch_b8)
      __builtin_prefetch(aRow + kd + 64, 0, 0);
      __builtin_prefetch(b0 + kd + 64, 0, 0);
    }
    BFPack bb;
    bb.h[0] = *(const u16x8*)(b0 + kd + g * 16);
    bb.h[1] = *(const u16x8*)(b0 + kd + g * 16 + 8);
    acc0 = __builtin_amdgcn_wmma_f32_16x16x32_bf16(false, a.v, false, bb.v, (short)0, acc0, false, false);
    bb.h[0] = *(const u16x8*)(b1 + kd + g * 16);
    bb.h[1] = *(const u16x8*)(b1 + kd + g * 16 + 8);
    acc1 = __builtin_amdgcn_wmma_f32_16x16x32_bf16(false, a.v, false, bb.v, (short)0, acc1, false, false);
    bb.h[0] = *(const u16x8*)(b2 + kd + g * 16);
    bb.h[1] = *(const u16x8*)(b2 + kd + g * 16 + 8);
    acc2 = __builtin_amdgcn_wmma_f32_16x16x32_bf16(false, a.v, false, bb.v, (short)0, acc2, false, false);
    bb.h[0] = *(const u16x8*)(b3 + kd + g * 16);
    bb.h[1] = *(const u16x8*)(b3 + kd + g * 16 + 8);
    acc3 = __builtin_amdgcn_wmma_f32_16x16x32_bf16(false, a.v, false, bb.v, (short)0, acc3, false, false);
  }

  v8f accs[4] = {acc0, acc1, acc2, acc3};
  for (int t = 0; t < 4; ++t) {
    const int eCol = eBase + t * 16 + ln;
    const float bv = bias[eCol];
    const int j  = eCol % 3;          // 0=q 1=k 2=v
    const int hd = eCol / 3;
    const int h  = hd >> 6;
    const int d  = hd & 63;
    for (int r = 0; r < 8; ++r) {
      const int m = mBase + r + 8 * g;
      const int b = m >> 11;
      const int n = m & (kN - 1);
      const float val = accs[t][r] + bv;
      const int bh = b * kH + h;
      if (j == 0)      qf[((size_t)bh * kN + n) * kD + d] = val;
      else if (j == 1) kf[((size_t)bh * kN + n) * kD + d] = val;
      else             vt16[((size_t)bh * kD + d) * kN + n] = f2bf(val); // transposed V
    }
  }
}

// ---------------------------------------------------------------- RMSNorm q/k -> bf16
__global__ void __launch_bounds__(256)
rms_cvt(const float* __restrict__ qf, const float* __restrict__ kf,
        const float* __restrict__ qw, const float* __restrict__ kw,
        u16* __restrict__ q16, u16* __restrict__ k16) {
  const int lane = threadIdx.x & 31;
  const int row = blockIdx.x * 8 + (threadIdx.x >> 5);
  const size_t base = (size_t)row * kD;
  const int d0 = lane * 2;
  {
    float x0 = qf[base + d0], x1 = qf[base + d0 + 1];
    float ss = x0 * x0 + x1 * x1;
    for (int m = 16; m; m >>= 1) ss += __shfl_xor(ss, m, 32);
    float inv = rsqrtf(ss * (1.0f / kD) + 1e-6f);
    unsigned int pk = (unsigned int)f2bf(x0 * inv * qw[d0]) |
                      ((unsigned int)f2bf(x1 * inv * qw[d0 + 1]) << 16);
    *(unsigned int*)&q16[base + d0] = pk;
  }
  {
    float x0 = kf[base + d0], x1 = kf[base + d0 + 1];
    float ss = x0 * x0 + x1 * x1;
    for (int m = 16; m; m >>= 1) ss += __shfl_xor(ss, m, 32);
    float inv = rsqrtf(ss * (1.0f / kD) + 1e-6f);
    unsigned int pk = (unsigned int)f2bf(x0 * inv * kw[d0]) |
                      ((unsigned int)f2bf(x1 * inv * kw[d0 + 1]) << 16);
    *(unsigned int*)&k16[base + d0] = pk;
  }
}

// ---------------------------------------------------------------- flash attention
// 8 waves/block share one (b,h); K/V tiles staged cooperatively in LDS
// (async-to-LDS when available), each wave owns a 16-query tile.
__global__ void __launch_bounds__(256)
attn(const u16* __restrict__ q16, const u16* __restrict__ k16,
     const u16* __restrict__ vt16, float* __restrict__ out) {
  __shared__ __align__(16) u16 Kl[32 * 64];     // 4KB: K[key][d], key-local 0..31
  __shared__ __align__(16) u16 Vl[64 * 32];     // 4KB: V^T[d][key], key-local 0..31
  __shared__ __align__(16) u16 Pl[8][16 * 32];  // per-wave P transpose tile
  const int tid  = threadIdx.x;
  const int lane = tid & 31;
  const int w    = tid >> 5;
  const int gw = blockIdx.x * 8 + w;            // 0..4095
  const int qt = gw & 127;
  const int bh = gw >> 7;                       // same for all 8 waves in block
  const int b = bh >> 4, h = bh & 15;
  const int g = lane >> 4, ln = lane & 15;
  const int qBase = qt * 16;

  const u16* kbase = k16 + (size_t)bh * kN * kD;
  const u16* vbase = vt16 + (size_t)bh * kD * kN;
  const int vrow = tid >> 2;                    // staging: V row (d), 64 rows
  const int vchk = (tid & 3) * 8;               // 4 x 8 u16 chunks per row

  // Q A-tiles (16x64 = two 16x32), resident
  const u16* qp = q16 + ((size_t)bh * kN + qBase + ln) * kD;
  BFPack aq0, aq1;
  aq0.h[0] = *(const u16x8*)(qp + g * 16);
  aq0.h[1] = *(const u16x8*)(qp + g * 16 + 8);
  aq1.h[0] = *(const u16x8*)(qp + 32 + g * 16);
  aq1.h[1] = *(const u16x8*)(qp + 32 + g * 16 + 8);

  v8f o0 = {}, o1 = {}, o2 = {}, o3 = {};
  float mrow[8], lrow[8];
  for (int r = 0; r < 8; ++r) { mrow[r] = -3.0e38f; lrow[r] = 0.0f; }

  for (int kc = 0; kc < kN; kc += 32) {
    __syncthreads();   // previous iteration's LDS readers done
    // ---- cooperative staging: K tile is one contiguous 4KB block; V rows 64B
#if HAVE_ASYNC_LDS
    async_cp16(kbase + (size_t)kc * kD + tid * 8, &Kl[tid * 8]);
    async_cp16(vbase + (size_t)vrow * kN + kc + vchk, &Vl[vrow * 32 + vchk]);
    asm volatile("s_wait_asynccnt 0" ::: "memory");
#else
    *(u16x8*)&Kl[tid * 8] = *(const u16x8*)(kbase + (size_t)kc * kD + tid * 8);
    *(u16x8*)&Vl[vrow * 32 + vchk] =
        *(const u16x8*)(vbase + (size_t)vrow * kN + kc + vchk);
#endif
    __syncthreads();   // tiles visible to all waves

    // ---- S = Q (16x64) @ K^T (64x32) from LDS
    v8f s0 = {}, s1 = {};
    {
      const u16* kl0 = &Kl[(size_t)ln * 64];          // key = ln      (ntile 0)
      const u16* kl1 = &Kl[(size_t)(16 + ln) * 64];   // key = 16+ln   (ntile 1)
      BFPack b00, b01, b10, b11;
      b00.h[0] = *(const u16x8*)(kl0 + g * 16);
      b00.h[1] = *(const u16x8*)(kl0 + g * 16 + 8);
      b01.h[0] = *(const u16x8*)(kl0 + 32 + g * 16);
      b01.h[1] = *(const u16x8*)(kl0 + 32 + g * 16 + 8);
      b10.h[0] = *(const u16x8*)(kl1 + g * 16);
      b10.h[1] = *(const u16x8*)(kl1 + g * 16 + 8);
      b11.h[0] = *(const u16x8*)(kl1 + 32 + g * 16);
      b11.h[1] = *(const u16x8*)(kl1 + 32 + g * 16 + 8);
      s0 = __builtin_amdgcn_wmma_f32_16x16x32_bf16(false, aq0.v, false, b00.v, (short)0, s0, false, false);
      s0 = __builtin_amdgcn_wmma_f32_16x16x32_bf16(false, aq1.v, false, b01.v, (short)0, s0, false, false);
      s1 = __builtin_amdgcn_wmma_f32_16x16x32_bf16(false, aq0.v, false, b10.v, (short)0, s1, false, false);
      s1 = __builtin_amdgcn_wmma_f32_16x16x32_bf16(false, aq1.v, false, b11.v, (short)0, s1, false, false);
    }
    // ---- online softmax (row = r + 8*g in 16-lane groups)
    float alpha[8];
    for (int r = 0; r < 8; ++r) {
      float a = s0[r] * 0.125f;    // 1/sqrt(64)
      float c = s1[r] * 0.125f;
      float mx = fmaxf(a, c);
      mx = fmaxf(mx, __shfl_xor(mx, 1, 32));
      mx = fmaxf(mx, __shfl_xor(mx, 2, 32));
      mx = fmaxf(mx, __shfl_xor(mx, 4, 32));
      mx = fmaxf(mx, __shfl_xor(mx, 8, 32));
      float mn = fmaxf(mrow[r], mx);
      alpha[r] = __expf(mrow[r] - mn);
      mrow[r] = mn;
      float p0 = __expf(a - mn);
      float p1 = __expf(c - mn);
      s0[r] = p0; s1[r] = p1;
      float sum = p0 + p1;
      sum += __shfl_xor(sum, 1, 32);
      sum += __shfl_xor(sum, 2, 32);
      sum += __shfl_xor(sum, 4, 32);
      sum += __shfl_xor(sum, 8, 32);
      lrow[r] = lrow[r] * alpha[r] + sum;
    }
    // ---- transpose P (C layout -> A layout) via per-wave LDS tile
    for (int r = 0; r < 8; ++r) {
      int row = r + 8 * g;
      Pl[w][row * 32 + ln]      = f2bf(s0[r]);
      Pl[w][row * 32 + 16 + ln] = f2bf(s1[r]);
    }
    __syncthreads();
    BFPack ap;
    ap.h[0] = *(const u16x8*)&Pl[w][ln * 32 + g * 16];
    ap.h[1] = *(const u16x8*)&Pl[w][ln * 32 + g * 16 + 8];
    // ---- rescale accumulators, O += P (16x32) @ V (32x64) from LDS
    for (int r = 0; r < 8; ++r) {
      o0[r] *= alpha[r]; o1[r] *= alpha[r]; o2[r] *= alpha[r]; o3[r] *= alpha[r];
    }
    {
      BFPack bv;
      const u16* vl = &Vl[(size_t)ln * 32 + g * 16];
      bv.h[0] = *(const u16x8*)(vl); bv.h[1] = *(const u16x8*)(vl + 8);
      o0 = __builtin_amdgcn_wmma_f32_16x16x32_bf16(false, ap.v, false, bv.v, (short)0, o0, false, false);
      vl += 16 * 32;
      bv.h[0] = *(const u16x8*)(vl); bv.h[1] = *(const u16x8*)(vl + 8);
      o1 = __builtin_amdgcn_wmma_f32_16x16x32_bf16(false, ap.v, false, bv.v, (short)0, o1, false, false);
      vl += 16 * 32;
      bv.h[0] = *(const u16x8*)(vl); bv.h[1] = *(const u16x8*)(vl + 8);
      o2 = __builtin_amdgcn_wmma_f32_16x16x32_bf16(false, ap.v, false, bv.v, (short)0, o2, false, false);
      vl += 16 * 32;
      bv.h[0] = *(const u16x8*)(vl); bv.h[1] = *(const u16x8*)(vl + 8);
      o3 = __builtin_amdgcn_wmma_f32_16x16x32_bf16(false, ap.v, false, bv.v, (short)0, o3, false, false);
    }
  }
  // ---- epilogue
  for (int r = 0; r < 8; ++r) {
    float inv = 1.0f / lrow[r];
    int row = qBase + r + 8 * g;
    size_t ob = ((size_t)b * kN + row) * kDim + h * kD + ln;
    out[ob]      = o0[r] * inv;
    out[ob + 16] = o1[r] * inv;
    out[ob + 32] = o2[r] * inv;
    out[ob + 48] = o3[r] * inv;
  }
}

// ---------------------------------------------------------------- launcher
extern "C" void kernel_launch(void* const* d_in, const int* in_sizes, int n_in,
                              void* d_out, int out_size, void* d_ws, size_t ws_size,
                              hipStream_t stream) {
  const float* x     = (const float*)d_in[0];
  const float* qkv_w = (const float*)d_in[1];
  const float* qkv_b = (const float*)d_in[2];
  const float* qnw   = (const float*)d_in[3];
  const float* knw   = (const float*)d_in[4];
  float* out = (float*)d_out;

  char* ws = (char*)d_ws;
  size_t off = 0;
  u16*   x16  = (u16*)(ws + off); off += (size_t)kB * kN * kDim * 2;      //  8 MB
  u16*   w16  = (u16*)(ws + off); off += (size_t)kE * kDim * 2;           //  6 MB
  float* qf   = (float*)(ws + off); off += (size_t)kB * kH * kN * kD * 4; // 16 MB
  float* kf   = (float*)(ws + off); off += (size_t)kB * kH * kN * kD * 4; // 16 MB
  u16*   q16  = (u16*)(ws + off); off += (size_t)kB * kH * kN * kD * 2;   //  8 MB
  u16*   k16  = (u16*)(ws + off); off += (size_t)kB * kH * kN * kD * 2;   //  8 MB
  u16*   vt16 = (u16*)(ws + off); off += (size_t)kB * kH * kN * kD * 2;   //  8 MB

  cvt_bf16<<<1024, 256, 0, stream>>>(x, x16, kB * kN * kDim);
  cvt_bf16<<<1024, 256, 0, stream>>>(qkv_w, w16, kE * kDim);

  // 256 M-tiles x 48 E-groups (16x64 per wave), 8 waves per block
  qkv_gemm<<<(256 * 48) / 8, 256, 0, stream>>>(x16, w16, qkv_b, qf, kf, vt16);

  // one wave per (b,h,n) row
  rms_cvt<<<(kB * kH * kN) / 8, 256, 0, stream>>>(qf, kf, qnw, knw, q16, k16);

  // one wave per 16-query tile, 8 waves/block sharing K/V staging
  attn<<<(kB * kH * (kN / 16)) / 8, 256, 0, stream>>>(q16, k16, vt16, out);
}
